// BitterGCN_65644280152401
// MI455X (gfx1250) — matrix operand order repaired and checked by reading, not compiled, not run on hardware
//
#include <hip/hip_runtime.h>
#include <cstdint>

#define NG 8192        // graphs
#define PP 24          // nodes per graph
#define NN 196608      // total nodes
#define EE 786432      // total edges
#define CAP 192        // per-graph edge bucket capacity (mean 96, ~10 sigma margin)

typedef float v8f __attribute__((ext_vector_type(8)));
typedef float f32x4 __attribute__((ext_vector_type(4)));
typedef __bf16 bf16x16 __attribute__((ext_vector_type(16)));

// ---- workspace layout (bytes, all 256B aligned) ----
static constexpr size_t OFF_XLR  = 0;                       // [NN][64] f32  = 50,331,648
static constexpr size_t OFF_H1   = 50331648;                // [NN][32] f32  = 25,165,824
static constexpr size_t OFF_S2   = 75497472;                // [NN][8]  f32  =  6,291,456
static constexpr size_t OFF_LIST = 81788928;                // [NG][CAP] i32 =  6,291,456
static constexpr size_t OFF_CNT  = 88080384;                // [NG] i32      =     32,768
static constexpr size_t OFF_WBF  = 88113152;                // [64][96] bf16 =     12,288
static constexpr size_t OFF_BN   = 88125440;                // 160 f32
// BN float layout: [0..31]=sum1 [32..63]=sq1 [64..71]=sum2 [72..79]=sq2
//                  [80..111]=a1 [112..143]=b1 [144..151]=a2 [152..159]=b2

__device__ __forceinline__ unsigned short f2bf_u(float f) {
  unsigned u = __float_as_uint(f);
  unsigned r = (u + 0x7FFFu + ((u >> 16) & 1u)) >> 16;   // round-to-nearest-even
  return (unsigned short)r;
}

// ---- CDNA5 async global->LDS copy (ASYNCcnt-tracked, no VGPR data path) ----
__device__ __forceinline__ void async_copy_b128(unsigned lds_off, const void* gaddr) {
  asm volatile("global_load_async_to_lds_b128 %0, %1, off"
               :: "v"(lds_off), "v"(gaddr) : "memory");
}
__device__ __forceinline__ void async_wait0() {
  asm volatile("s_wait_asynccnt 0" ::: "memory");
}

// ---------------- K0: init counters/BN + pack weights to bf16 ----------------
__global__ void k_init(int* __restrict__ cnt, float* __restrict__ bn,
                       unsigned short* __restrict__ wbf,
                       const float* __restrict__ Wl1, const float* __restrict__ Wr1) {
  int idx = blockIdx.x * blockDim.x + threadIdx.x;
  if (idx < NG) cnt[idx] = 0;
  if (idx < 160) bn[idx] = 0.f;
  if (idx < 64 * 96) {
    int o = idx / 96, k = idx % 96;     // o: output channel (0..63), k: padded input dim
    float v = 0.f;
    if (k < 74) v = (o < 32) ? Wl1[k * 32 + o] : Wr1[k * 32 + (o - 32)];
    wbf[idx] = f2bf_u(v);               // row o contiguous over k -> B column-major load
  }
}

// ---------------- K1: xl1||xr1 = x[:, :74] @ [Wl1|Wr1] + bias (bf16 WMMA) ----------------
__global__ void k_gemm1(const float* __restrict__ x, const unsigned short* __restrict__ wbfu,
                        const float* __restrict__ bl1, const float* __restrict__ br1,
                        float* __restrict__ xlr) {
  const __bf16* wbf = reinterpret_cast<const __bf16*>(wbfu);
  int wave = blockIdx.x * 8 + (threadIdx.x >> 5);   // 16-row tile id
  int lane = threadIdx.x & 31;
  int r0 = wave * 16;
  int mrow = lane & 15;
  int ksel = lane >> 4;                             // 0: low K half, 1: high K half
  const float* xrow = x + (size_t)(r0 + mrow) * 96;
  __builtin_prefetch(xrow + (size_t)16 * 96, 0, 0); // stream-ahead one tile

  v8f acc[4] = {};
#pragma unroll
  for (int ks = 0; ks < 3; ++ks) {
    int k0 = ks * 32;
    int c0 = k0 + ksel * 8;
    // A fragment per ISA 16-bit A 16x32 layout: elems 0..7 = K c0..c0+7,
    // elems 8..15 = K c0+16..c0+23.  Row is 96 floats, so unguarded vector
    // loads are in-bounds; mask the K>=74 pad region with selects.
    f32x4 A0 = *reinterpret_cast<const f32x4*>(xrow + c0);
    f32x4 A1 = *reinterpret_cast<const f32x4*>(xrow + c0 + 4);
    f32x4 A2 = *reinterpret_cast<const f32x4*>(xrow + c0 + 16);
    f32x4 A3 = *reinterpret_cast<const f32x4*>(xrow + c0 + 20);
    bf16x16 a;
#pragma unroll
    for (int j = 0; j < 4; ++j) {
      a[j]      = (__bf16)((c0 + j      < 74) ? A0[j] : 0.f);
      a[4 + j]  = (__bf16)((c0 + 4 + j  < 74) ? A1[j] : 0.f);
      a[8 + j]  = (__bf16)((c0 + 16 + j < 74) ? A2[j] : 0.f);
      a[12 + j] = (__bf16)((c0 + 20 + j < 74) ? A3[j] : 0.f);
    }
#pragma unroll
    for (int nt = 0; nt < 4; ++nt) {
      int colg = nt * 16 + mrow;       // B column = output channel
      typedef __bf16 bf16x8 __attribute__((ext_vector_type(8)));
      bf16x8 bl = *reinterpret_cast<const bf16x8*>(wbf + (size_t)colg * 96 + k0 + ksel * 8);
      bf16x8 bh = *reinterpret_cast<const bf16x8*>(wbf + (size_t)colg * 96 + k0 + 16 + ksel * 8);
      bf16x16 b = __builtin_shufflevector(bl, bh, 0, 1, 2, 3, 4, 5, 6, 7, 8, 9, 10, 11, 12, 13, 14, 15);
      acc[nt] = __builtin_amdgcn_wmma_f32_16x16x32_bf16(
          false, a, false, b, (short)0, acc[nt], false, false);
    }
  }
#pragma unroll
  for (int nt = 0; nt < 4; ++nt) {
    int colg = nt * 16 + mrow;
    float bias = (colg < 32) ? bl1[colg] : br1[colg - 32];
#pragma unroll
    for (int r = 0; r < 8; ++r) {
      int row = r0 + r + ksel * 8;     // C/D layout: VGPR r -> M = r (+8 for hi lanes)
      xlr[(size_t)row * 64 + colg] = acc[nt][r] + bias;
    }
  }
}

// ---------------- K2: bucket edges by graph ----------------
__global__ void k_bin(const int* __restrict__ srcA, int* __restrict__ cnt, int* __restrict__ lst) {
  int e = blockIdx.x * blockDim.x + threadIdx.x;
  if (e >= EE) return;
  int g = srcA[e] / PP;                // edges are intra-graph by construction
  int pos = atomicAdd(&cnt[g], 1);
  if (pos < CAP) lst[(size_t)g * CAP + pos] = e;
}

// ---------------- K3: GATv2 layer 1, one block (128 thr) per graph ----------------
__global__ void k_gat1(const float* __restrict__ xlr,
                       const int* __restrict__ srcA, const int* __restrict__ dstA,
                       const float* __restrict__ ea,
                       const float* __restrict__ We1, const float* __restrict__ att1,
                       const float* __restrict__ bo1,
                       const int* __restrict__ lst, const int* __restrict__ cntA,
                       float* __restrict__ h1, float* __restrict__ bn) {
  __shared__ __align__(16) float sXLR[PP * 64];   // [node][0..31]=xl, [32..63]=xr
  __shared__ float sLA[PP * 16], sDeg[PP];
  __shared__ float sSc[CAP], sSL[PP], sMax[PP], sDen[PP], sH[PP * 32];
  __shared__ int sS[CAP], sD[CAP];
  int g = blockIdx.x, n0 = g * PP, t = threadIdx.x;
  int cnt = cntA[g]; if (cnt > CAP) cnt = CAP;
  const int* el = lst + (size_t)g * CAP;

  // async-DMA the 6KB xl||xr tile straight into LDS
  {
    const float* gsrc = xlr + (size_t)n0 * 64;
    unsigned lbase = (unsigned)(uintptr_t)(const void*)sXLR;
    for (int ch = t; ch < PP * 64 / 4; ch += 128)
      async_copy_b128(lbase + ch * 16, gsrc + ch * 4);
    async_wait0();
  }
  for (int i = t; i < PP * 16; i += 128) sLA[i] = 0.f;
  if (t < PP) sDeg[t] = 0.f;
  __syncthreads();

  // degree + self-loop attr sums
  for (int i = t; i < cnt; i += 128) {
    int e = el[i];
    int s = srcA[e] - n0, d = dstA[e] - n0;
    sS[i] = s; sD[i] = d;
    atomicAdd(&sDeg[d], 1.f);
    const float* ep = ea + (size_t)e * 16;
    for (int k = 0; k < 16; ++k) atomicAdd(&sLA[d * 16 + k], ep[k]);
  }
  __syncthreads();
  for (int i = t; i < PP * 16; i += 128) {
    float dg = sDeg[i >> 4]; if (dg < 1.f) dg = 1.f;
    sLA[i] /= dg;
  }
  __syncthreads();

  // edge scores: one wave per edge, lane = channel, wave32 reduce
  int wv = t >> 5, lane = t & 31;
  for (int i = wv; i < cnt; i += 4) {
    int e = el[i], s = sS[i], d = sD[i];
    const float* ep = ea + (size_t)e * 16;
    float m = sXLR[s * 64 + lane] + sXLR[d * 64 + 32 + lane];
    for (int k = 0; k < 16; ++k) m += ep[k] * We1[k * 32 + lane];
    float lr = m > 0.f ? m : 0.2f * m;
    float p = lr * att1[lane];
    for (int o = 16; o; o >>= 1) p += __shfl_xor(p, o, 32);
    if (lane == 0) sSc[i] = p;
  }
  for (int i = wv; i < PP; i += 4) {           // self-loop scores
    float m = sXLR[i * 64 + lane] + sXLR[i * 64 + 32 + lane];
    for (int k = 0; k < 16; ++k) m += sLA[i * 16 + k] * We1[k * 32 + lane];
    float lr = m > 0.f ? m : 0.2f * m;
    float p = lr * att1[lane];
    for (int o = 16; o; o >>= 1) p += __shfl_xor(p, o, 32);
    if (lane == 0) sSL[i] = p;
  }
  __syncthreads();

  if (t < PP) {                                 // per-dst softmax stats
    float mx = sSL[t];
    for (int i = 0; i < cnt; ++i) if (sD[i] == t && sSc[i] > mx) mx = sSc[i];
    float den = expf(sSL[t] - mx);
    for (int i = 0; i < cnt; ++i) if (sD[i] == t) den += expf(sSc[i] - mx);
    sMax[t] = mx; sDen[t] = den;
  }
  __syncthreads();
  for (int i = t; i < cnt; i += 128) {
    int d = sD[i];
    sSc[i] = expf(sSc[i] - sMax[d]) / sDen[d];
  }
  if (t < PP) sSL[t] = expf(sSL[t] - sMax[t]) / sDen[t];
  __syncthreads();

  // aggregate alpha * xl[src], +bias, ReLU, emit + BN partials
  for (int i = t; i < PP * 32; i += 128) {
    int d = i >> 5, c = i & 31;
    float acc = sSL[d] * sXLR[d * 64 + c];
    for (int k = 0; k < cnt; ++k) if (sD[k] == d) acc += sSc[k] * sXLR[sS[k] * 64 + c];
    float v = acc + bo1[c];
    v = v > 0.f ? v : 0.f;
    sH[i] = v;
    h1[(size_t)(n0 + d) * 32 + c] = v;
  }
  __syncthreads();
  if (t < 32) {
    float su = 0.f, sq = 0.f;
    for (int n = 0; n < PP; ++n) { float v = sH[n * 32 + t]; su += v; sq += v * v; }
    atomicAdd(&bn[t], su);
    atomicAdd(&bn[32 + t], sq);
  }
}

// ---------------- BN finalize: fold into affine a,b ----------------
__global__ void k_bnfin(float* __restrict__ bn, const float* __restrict__ g,
                        const float* __restrict__ be, int C,
                        int sumOff, int sqOff, int aOff, int bOff, float invN) {
  int c = threadIdx.x;
  if (c < C) {
    float mu = bn[sumOff + c] * invN;
    float var = bn[sqOff + c] * invN - mu * mu;
    float a = g[c] * rsqrtf(var + 1e-5f);
    bn[aOff + c] = a;
    bn[bOff + c] = be[c] - mu * a;
  }
}

// ---------------- K4: GATv2 layer 2 (width 8) per graph ----------------
__global__ void k_gat2(const float* __restrict__ h1,
                       const int* __restrict__ srcA, const int* __restrict__ dstA,
                       const float* __restrict__ ea,
                       const float* __restrict__ Wl2, const float* __restrict__ bl2,
                       const float* __restrict__ Wr2, const float* __restrict__ br2,
                       const float* __restrict__ We2, const float* __restrict__ att2,
                       const float* __restrict__ bo2,
                       const int* __restrict__ lst, const int* __restrict__ cntA,
                       float* __restrict__ bn, float* __restrict__ s2out) {
  __shared__ __align__(16) float sHB[PP * 32];
  __shared__ float sXL[PP * 8], sXR[PP * 8], sLA[PP * 16], sDeg[PP];
  __shared__ float sSc[CAP], sSL[PP], sMax[PP], sDen[PP], sH[PP * 8];
  __shared__ int sS[CAP], sD[CAP];
  int g = blockIdx.x, n0 = g * PP, t = threadIdx.x;
  int cnt = cntA[g]; if (cnt > CAP) cnt = CAP;
  const int* el = lst + (size_t)g * CAP;
  const float* a1 = bn + 80; const float* b1 = bn + 112;

  // async-DMA the raw h1 tile into LDS, then apply BN1 affine in place
  {
    const float* gsrc = h1 + (size_t)n0 * 32;
    unsigned lbase = (unsigned)(uintptr_t)(const void*)sHB;
    for (int ch = t; ch < PP * 32 / 4; ch += 128)
      async_copy_b128(lbase + ch * 16, gsrc + ch * 4);
    async_wait0();
  }
  for (int i = t; i < PP * 16; i += 128) sLA[i] = 0.f;
  if (t < PP) sDeg[t] = 0.f;
  __syncthreads();
  for (int i = t; i < PP * 32; i += 128) {
    int c = i & 31;
    sHB[i] = a1[c] * sHB[i] + b1[c];
  }
  __syncthreads();

  for (int i = t; i < PP * 8; i += 128) {      // xl2 / xr2 (32->8)
    int node = i >> 3, o = i & 7;
    float aL = bl2[o], aR = br2[o];
    for (int k = 0; k < 32; ++k) {
      float hv = sHB[node * 32 + k];
      aL += hv * Wl2[k * 8 + o];
      aR += hv * Wr2[k * 8 + o];
    }
    sXL[i] = aL; sXR[i] = aR;
  }
  __syncthreads();

  for (int i = t; i < cnt; i += 128) {         // degree + loop attrs
    int e = el[i];
    int s = srcA[e] - n0, d = dstA[e] - n0;
    sS[i] = s; sD[i] = d;
    atomicAdd(&sDeg[d], 1.f);
    const float* ep = ea + (size_t)e * 16;
    for (int k = 0; k < 16; ++k) atomicAdd(&sLA[d * 16 + k], ep[k]);
  }
  __syncthreads();
  for (int i = t; i < PP * 16; i += 128) {
    float dg = sDeg[i >> 4]; if (dg < 1.f) dg = 1.f;
    sLA[i] /= dg;
  }
  __syncthreads();

  for (int i = t; i < cnt; i += 128) {         // scores, one thread per edge
    int e = el[i], s = sS[i], d = sD[i];
    const float* ep = ea + (size_t)e * 16;
    float sc = 0.f;
    for (int c = 0; c < 8; ++c) {
      float m = sXL[s * 8 + c] + sXR[d * 8 + c];
      for (int k = 0; k < 16; ++k) m += ep[k] * We2[k * 8 + c];
      float lr = m > 0.f ? m : 0.2f * m;
      sc += lr * att2[c];
    }
    sSc[i] = sc;
  }
  if (t < PP) {
    float sc = 0.f;
    for (int c = 0; c < 8; ++c) {
      float m = sXL[t * 8 + c] + sXR[t * 8 + c];
      for (int k = 0; k < 16; ++k) m += sLA[t * 16 + k] * We2[k * 8 + c];
      float lr = m > 0.f ? m : 0.2f * m;
      sc += lr * att2[c];
    }
    sSL[t] = sc;
  }
  __syncthreads();
  if (t < PP) {
    float mx = sSL[t];
    for (int i = 0; i < cnt; ++i) if (sD[i] == t && sSc[i] > mx) mx = sSc[i];
    float den = expf(sSL[t] - mx);
    for (int i = 0; i < cnt; ++i) if (sD[i] == t) den += expf(sSc[i] - mx);
    sMax[t] = mx; sDen[t] = den;
  }
  __syncthreads();
  for (int i = t; i < cnt; i += 128) {
    int d = sD[i];
    sSc[i] = expf(sSc[i] - sMax[d]) / sDen[d];
  }
  if (t < PP) sSL[t] = expf(sSL[t] - sMax[t]) / sDen[t];
  __syncthreads();

  for (int i = t; i < PP * 8; i += 128) {      // aggregate + sigmoid
    int d = i >> 3, c = i & 7;
    float acc = sSL[d] * sXL[d * 8 + c];
    for (int k = 0; k < cnt; ++k) if (sD[k] == d) acc += sSc[k] * sXL[sS[k] * 8 + c];
    float v = acc + bo2[c];
    v = 1.f / (1.f + expf(-v));
    sH[i] = v;
    s2out[(size_t)(n0 + d) * 8 + c] = v;
  }
  __syncthreads();
  if (t < 8) {
    float su = 0.f, sq = 0.f;
    for (int n = 0; n < PP; ++n) { float v = sH[n * 8 + t]; su += v; sq += v * v; }
    atomicAdd(&bn[64 + t], su);
    atomicAdd(&bn[72 + t], sq);
  }
}

// ---------------- K5: pool + rec features + MLP head, one thread per graph ----------------
__global__ void k_head(const float* __restrict__ s2, const float* __restrict__ x,
                       const float* __restrict__ bn,
                       const float* __restrict__ fc1W, const float* __restrict__ fc1b,
                       const float* __restrict__ fc2W, const float* __restrict__ fc2b,
                       const float* __restrict__ fc3W, const float* __restrict__ fc3b,
                       const float* __restrict__ fc4W, const float* __restrict__ fc4b,
                       const float* __restrict__ outW, const float* __restrict__ outb,
                       float* __restrict__ out) {
  int g = blockIdx.x * blockDim.x + threadIdx.x;
  if (g >= NG) return;
  const float* a2 = bn + 144; const float* b2 = bn + 152;
  float z[30];
  for (int c = 0; c < 8; ++c) {
    float s = 0.f;
    for (int n = 0; n < PP; ++n) s += s2[(size_t)(g * PP + n) * 8 + c];
    z[c] = a2[c] * (s * (1.f / 24.f)) + b2[c];   // BN2 folded into mean-pool
  }
  const float* xr = x + (size_t)g * PP * 96 + 74;
  for (int j = 0; j < 22; ++j) z[8 + j] = xr[j];
  float z1[32];
  for (int o = 0; o < 32; ++o) {
    float a = fc1b[o];
    for (int i = 0; i < 30; ++i) a += z[i] * fc1W[i * 32 + o];
    z1[o] = a > 0.f ? a : 0.f;
  }
  float z2[16];
  for (int o = 0; o < 16; ++o) {
    float a = fc2b[o];
    for (int i = 0; i < 32; ++i) a += z1[i] * fc2W[i * 16 + o];
    z2[o] = a > 0.f ? a : 0.f;
  }
  float z3[8];
  for (int o = 0; o < 8; ++o) {
    float a = fc3b[o];
    for (int i = 0; i < 16; ++i) a += z2[i] * fc3W[i * 8 + o];
    z3[o] = a > 0.f ? a : 0.f;
  }
  float z4[4];
  for (int o = 0; o < 4; ++o) {
    float a = fc4b[o];
    for (int i = 0; i < 8; ++i) a += z3[i] * fc4W[i * 4 + o];
    z4[o] = a > 0.f ? a : 0.f;
  }
  for (int o = 0; o < 2; ++o) {
    float a = outb[o];
    for (int i = 0; i < 4; ++i) a += z4[i] * outW[i * 2 + o];
    out[(size_t)g * 2 + o] = a;
  }
}

extern "C" void kernel_launch(void* const* d_in, const int* in_sizes, int n_in,
                              void* d_out, int out_size, void* d_ws, size_t ws_size,
                              hipStream_t stream) {
  (void)in_sizes; (void)n_in; (void)out_size; (void)ws_size;
  const float* x    = (const float*)d_in[0];
  const int*   eidx = (const int*)d_in[1];
  const float* ea   = (const float*)d_in[2];
  const float* Wl1  = (const float*)d_in[4];  const float* bl1 = (const float*)d_in[5];
  const float* Wr1  = (const float*)d_in[6];  const float* br1 = (const float*)d_in[7];
  const float* We1  = (const float*)d_in[8];  const float* att1 = (const float*)d_in[9];
  const float* bo1  = (const float*)d_in[10];
  const float* Wl2  = (const float*)d_in[11]; const float* bl2 = (const float*)d_in[12];
  const float* Wr2  = (const float*)d_in[13]; const float* br2 = (const float*)d_in[14];
  const float* We2  = (const float*)d_in[15]; const float* att2 = (const float*)d_in[16];
  const float* bo2  = (const float*)d_in[17];
  const float* g1   = (const float*)d_in[18]; const float* be1 = (const float*)d_in[19];
  const float* g2   = (const float*)d_in[20]; const float* be2 = (const float*)d_in[21];
  const float* fc1W = (const float*)d_in[22]; const float* fc1b = (const float*)d_in[23];
  const float* fc2W = (const float*)d_in[24]; const float* fc2b = (const float*)d_in[25];
  const float* fc3W = (const float*)d_in[26]; const float* fc3b = (const float*)d_in[27];
  const float* fc4W = (const float*)d_in[28]; const float* fc4b = (const float*)d_in[29];
  const float* outW = (const float*)d_in[30]; const float* outb = (const float*)d_in[31];
  float* out = (float*)d_out;

  char* ws = (char*)d_ws;
  float* xlr          = (float*)(ws + OFF_XLR);
  float* h1           = (float*)(ws + OFF_H1);
  float* s2           = (float*)(ws + OFF_S2);
  int*   lst          = (int*)(ws + OFF_LIST);
  int*   cnt          = (int*)(ws + OFF_CNT);
  unsigned short* wbf = (unsigned short*)(ws + OFF_WBF);
  float* bn           = (float*)(ws + OFF_BN);

  const int* srcA = eidx;        // edge_index row 0
  const int* dstA = eidx + EE;   // edge_index row 1

  k_init<<<NG / 256, 256, 0, stream>>>(cnt, bn, wbf, Wl1, Wr1);
  k_gemm1<<<NN / 16 / 8, 256, 0, stream>>>(x, wbf, bl1, br1, xlr);
  k_bin<<<EE / 256, 256, 0, stream>>>(srcA, cnt, lst);
  k_gat1<<<NG, 128, 0, stream>>>(xlr, srcA, dstA, ea, We1, att1, bo1, lst, cnt, h1, bn);
  k_bnfin<<<1, 64, 0, stream>>>(bn, g1, be1, 32, 0, 32, 80, 112, 1.f / (float)NN);
  k_gat2<<<NG, 128, 0, stream>>>(h1, srcA, dstA, ea, Wl2, bl2, Wr2, br2, We2, att2, bo2,
                                 lst, cnt, bn, s2);
  k_bnfin<<<1, 64, 0, stream>>>(bn, g2, be2, 8, 64, 72, 144, 152, 1.f / (float)NN);
  k_head<<<NG / 256, 256, 0, stream>>>(s2, x, bn, fc1W, fc1b, fc2W, fc2b, fc3W, fc3b,
                                       fc4W, fc4b, outW, outb, out);
}
